// GatedGDTLayer_1769526526467
// MI455X (gfx1250) — matrix operand order repaired and checked by reading, not compiled, not run on hardware
//
#include <hip/hip_runtime.h>
#include <hip/hip_bf16.h>
#include <math.h>

#define NNODES 40000
#define DEG    16
#define NEDGES (NNODES * DEG)
#define DIN    128
#define NH     8
#define HD     16
#define FFH    512
#define ALPHA  0.15f
// log(16)/16  (every node has exactly 16 in-edges by construction)
#define LOGDEG_OVER_D 0.17328679513998632f

typedef __attribute__((ext_vector_type(16))) _Float16 v16h;
typedef __attribute__((ext_vector_type(8)))  _Float16 v8h;
typedef __attribute__((ext_vector_type(4)))  _Float16 v4h;
typedef __attribute__((ext_vector_type(8)))  float    v8f;

// ---------------------------------------------------------------------------
// WMMA fragment loaders (CDNA5 16-bit layouts, CDNA5 ISA §7.12.2)
// A: 16x32 (MxK). lane r=lane&15 row, hi=lane>>4: elems0..7 = K k0+hi*8..+7,
//    elems8..15 = K k0+16+hi*8..+7  -> two 16B loads.
// B: 32x16 (KxN) = W^T tile, W row-major [nout][K]. lane c=lane&15 col,
//    elems0..7 = K k0+hi*8..+7, elems8..15 = K k0+16+hi*8..+7 -> two 16B loads.
// C/D: 16x16 f32: vgpr v -> row m0+hi*8+v, col o0+(lane&15).
// ---------------------------------------------------------------------------
__device__ inline v16h load_a_frag(const _Float16* __restrict__ A, int ldk,
                                   int m0, int k0, int lane) {
  int r = lane & 15, hi = lane >> 4;
  const _Float16* row = A + (size_t)(m0 + r) * ldk + k0 + hi * 8;
  union { v16h v; v8h h[2]; } u;
  u.h[0] = *(const v8h*)(row);
  u.h[1] = *(const v8h*)(row + 16);
  return u.v;
}

__device__ inline v16h load_b_frag(const _Float16* __restrict__ W, int ldk,
                                   int o0, int k0, int lane) {
  int c = lane & 15, hi = lane >> 4;
  const _Float16* row = W + (size_t)(o0 + c) * ldk + k0 + hi * 8;
  union { v16h v; v8h h[2]; } u;
  u.h[0] = *(const v8h*)(row);
  u.h[1] = *(const v8h*)(row + 16);
  return u.v;
}

__device__ inline float wave_sum32(float v) {
#pragma unroll
  for (int m = 16; m > 0; m >>= 1) v += __shfl_xor(v, m, 32);
  return v;
}

// ---------------------------------------------------------------------------
// f32 -> f16 convert (weights)
// ---------------------------------------------------------------------------
__global__ void k_cvt_f16(const float* __restrict__ in, _Float16* __restrict__ out, int n) {
  int i = blockIdx.x * 256 + threadIdx.x;
  if (i < n) out[i] = (_Float16)in[i];
}

// ---------------------------------------------------------------------------
// LayerNorm over 128 cols, one wave32 per row, output f16
// ---------------------------------------------------------------------------
__global__ void k_ln_f16(const float* __restrict__ x, const float* __restrict__ g,
                         const float* __restrict__ b, _Float16* __restrict__ out) {
  int wave = threadIdx.x >> 5, lane = threadIdx.x & 31;
  int row = blockIdx.x * 8 + wave;
  size_t base = (size_t)row * DIN + lane * 4;
  float4 v = *(const float4*)(x + base);
  float s  = v.x + v.y + v.z + v.w;
  float sq = v.x * v.x + v.y * v.y + v.z * v.z + v.w * v.w;
  s  = wave_sum32(s);
  sq = wave_sum32(sq);
  float mean = s * (1.0f / DIN);
  float var  = sq * (1.0f / DIN) - mean * mean;
  float inv  = rsqrtf(var + 1e-5f);
  int c = lane * 4;
  v4h o;
  o[0] = (_Float16)((v.x - mean) * inv * g[c + 0] + b[c + 0]);
  o[1] = (_Float16)((v.y - mean) * inv * g[c + 1] + b[c + 1]);
  o[2] = (_Float16)((v.z - mean) * inv * g[c + 2] + b[c + 2]);
  o[3] = (_Float16)((v.w - mean) * inv * g[c + 3] + b[c + 3]);
  *(v4h*)(out + base) = o;
}

// ---------------------------------------------------------------------------
// residual add + LayerNorm2: rst = f + feat (f32 out), x16 = f16(LN(rst))
// ---------------------------------------------------------------------------
__global__ void k_rst_ln(const float* __restrict__ f, const float* __restrict__ feat,
                         const float* __restrict__ g, const float* __restrict__ b,
                         float* __restrict__ rst, _Float16* __restrict__ x16) {
  int wave = threadIdx.x >> 5, lane = threadIdx.x & 31;
  int row = blockIdx.x * 8 + wave;
  size_t base = (size_t)row * DIN + lane * 4;
  float4 fv = *(const float4*)(f + base);
  float4 tv = *(const float4*)(feat + base);
  float4 r;
  r.x = fv.x + tv.x; r.y = fv.y + tv.y; r.z = fv.z + tv.z; r.w = fv.w + tv.w;
  *(float4*)(rst + base) = r;
  float s  = r.x + r.y + r.z + r.w;
  float sq = r.x * r.x + r.y * r.y + r.z * r.z + r.w * r.w;
  s  = wave_sum32(s);
  sq = wave_sum32(sq);
  float mean = s * (1.0f / DIN);
  float var  = sq * (1.0f / DIN) - mean * mean;
  float inv  = rsqrtf(var + 1e-5f);
  int c = lane * 4;
  v4h o;
  o[0] = (_Float16)((r.x - mean) * inv * g[c + 0] + b[c + 0]);
  o[1] = (_Float16)((r.y - mean) * inv * g[c + 1] + b[c + 1]);
  o[2] = (_Float16)((r.z - mean) * inv * g[c + 2] + b[c + 2]);
  o[3] = (_Float16)((r.w - mean) * inv * g[c + 3] + b[c + 3]);
  *(v4h*)(x16 + base) = o;
}

// ---------------------------------------------------------------------------
// Fused projection GEMM: out = h16 @ W^T for W in {Wh, Wt, We} (blockIdx.z)
// grid (N/16, 1, 3), block 256 (8 waves, wave w -> 16 out cols)
// ---------------------------------------------------------------------------
__global__ void k_gemm_proj(const _Float16* __restrict__ A,
                            const _Float16* __restrict__ Wh16,
                            const _Float16* __restrict__ Wt16,
                            const _Float16* __restrict__ We16,
                            float* __restrict__ fh, float* __restrict__ ft,
                            float* __restrict__ fe) {
  int wave = threadIdx.x >> 5, lane = threadIdx.x & 31;
  int m0 = blockIdx.x * 16;
  int o0 = wave * 16;
  const _Float16* W = (blockIdx.z == 0) ? Wh16 : (blockIdx.z == 1) ? Wt16 : We16;
  float* out = (blockIdx.z == 0) ? fh : (blockIdx.z == 1) ? ft : fe;
  v8f c = {};
#pragma unroll
  for (int k0 = 0; k0 < DIN; k0 += 32) {
    v16h a = load_a_frag(A, DIN, m0, k0, lane);
    v16h b = load_b_frag(W, DIN, o0, k0, lane);
    c = __builtin_amdgcn_wmma_f32_16x16x32_f16(false, a, false, b, (short)0, c,
                                               false, false);
  }
  int col = lane & 15, hi = lane >> 4;
#pragma unroll
  for (int v = 0; v < 8; ++v)
    out[(size_t)(m0 + hi * 8 + v) * DIN + o0 + col] = c[v];
}

// ---------------------------------------------------------------------------
// gate scores: gh[n,h] = <fh[n,h,:], g_head[h,:]>, gt similarly
// ---------------------------------------------------------------------------
__global__ void k_gate(const float* __restrict__ fh, const float* __restrict__ ft,
                       const float* __restrict__ g_head, const float* __restrict__ g_tail,
                       float* __restrict__ gh, float* __restrict__ gt) {
  int idx = blockIdx.x * 256 + threadIdx.x;           // n*NH + h
  if (idx >= NNODES * NH) return;
  int h = idx & 7;
  int n = idx >> 3;
  const float* p1 = fh + (size_t)n * DIN + h * HD;
  const float* p2 = ft + (size_t)n * DIN + h * HD;
  const float* w1 = g_head + h * HD;
  const float* w2 = g_tail + h * HD;
  float s1 = 0.f, s2 = 0.f;
#pragma unroll
  for (int d = 0; d < HD; ++d) { s1 += p1[d] * w1[d]; s2 += p2[d] * w2[d]; }
  gh[idx] = s1;
  gt[idx] = s2;
}

// ---------------------------------------------------------------------------
// edge attention + gated edge-softmax per (dst,head). Edges of node i are
// contiguous [i*16, i*16+16) -> width-16 shuffle reductions.
// block 256 -> 2 nodes; t%128: h = t/16, j = t%16 (16-lane groups wave-aligned)
// ---------------------------------------------------------------------------
__global__ void k_attn(const float* __restrict__ fh, const float* __restrict__ ft,
                       const float* __restrict__ attn_p, const float* __restrict__ gh,
                       const float* __restrict__ gt, const int* __restrict__ src,
                       float* __restrict__ attn) {
  int node = blockIdx.x * 2 + (threadIdx.x >> 7);
  int t = threadIdx.x & 127;
  int h = t >> 4, j = t & 15;
  int eidx = node * DEG + j;
  int s = src[eidx];
  const float* ph = fh + (size_t)s * DIN + h * HD;
  const float* pt = ft + (size_t)node * DIN + h * HD;
  const float* pa = attn_p + h * HD;
  float dot = 0.f;
#pragma unroll
  for (int d = 0; d < HD; ++d) dot += ph[d] * pt[d] * pa[d];
  float e = dot * LOGDEG_OVER_D;
  float gate = 1.0f / (1.0f + expf(-(gh[s * NH + h] + gt[node * NH + h])));
  float mx = e;
#pragma unroll
  for (int m = 8; m > 0; m >>= 1) mx = fmaxf(mx, __shfl_xor(mx, m, 16));
  float tv = expf(e - mx) * gate;
  float sm = tv;
#pragma unroll
  for (int m = 8; m > 0; m >>= 1) sm += __shfl_xor(sm, m, 16);
  attn[(size_t)eidx * NH + h] = tv / sm;
}

// ---------------------------------------------------------------------------
// one PPR diffusion hop: fn[i,c] = (1-a)*sum_j attn[i*16+j, c/16]*f[src_j, c] + a*f0[i,c]
// block 256 -> 2 nodes, 128 threads per node (c = h*16+d), coalesced gathers
// (f fits in L2; prefetch all 16 neighbor rows first to pipeline).
// ---------------------------------------------------------------------------
__global__ void k_diffuse(const float* __restrict__ f, const float* __restrict__ f0,
                          const float* __restrict__ attn, const int* __restrict__ src,
                          float* __restrict__ fn) {
  int node = blockIdx.x * 2 + (threadIdx.x >> 7);
  int c = threadIdx.x & 127;
  int h = c >> 4;
  size_t eb = (size_t)node * DEG;
  int sj[DEG];
#pragma unroll
  for (int j = 0; j < DEG; ++j) sj[j] = src[eb + j];
#pragma unroll
  for (int j = 0; j < DEG; ++j)
    __builtin_prefetch(f + (size_t)sj[j] * DIN + c, 0, 0);
  float acc = 0.f;
#pragma unroll
  for (int j = 0; j < DEG; ++j) {
    float w = attn[(eb + j) * NH + h];
    acc += w * f[(size_t)sj[j] * DIN + c];
  }
  size_t o = (size_t)node * DIN + c;
  fn[o] = (1.0f - ALPHA) * acc + ALPHA * f0[o];
}

// ---------------------------------------------------------------------------
// FFN layer 1: y16 = f16(relu(x16 @ W1^T + b1)), W1 [512,128]
// grid (N/16, 4), block 256; wave w -> out cols (by*8+w)*16
// ---------------------------------------------------------------------------
__global__ void k_ffn1(const _Float16* __restrict__ X, const _Float16* __restrict__ W1,
                       const float* __restrict__ b1, _Float16* __restrict__ Y) {
  int wave = threadIdx.x >> 5, lane = threadIdx.x & 31;
  int m0 = blockIdx.x * 16;
  int o0 = (blockIdx.y * 8 + wave) * 16;
  v8f c = {};
#pragma unroll
  for (int k0 = 0; k0 < DIN; k0 += 32) {
    v16h a = load_a_frag(X, DIN, m0, k0, lane);
    v16h b = load_b_frag(W1, DIN, o0, k0, lane);
    c = __builtin_amdgcn_wmma_f32_16x16x32_f16(false, a, false, b, (short)0, c,
                                               false, false);
  }
  int col = lane & 15, hi = lane >> 4;
  float bias = b1[o0 + col];
#pragma unroll
  for (int v = 0; v < 8; ++v) {
    float r = fmaxf(c[v] + bias, 0.0f);
    Y[(size_t)(m0 + hi * 8 + v) * FFH + o0 + col] = (_Float16)r;
  }
}

// ---------------------------------------------------------------------------
// FFN layer 2 + residual: out = y16 @ W2^T + b2 + rst, W2 [128,512]
// grid (N/16), block 256; wave w -> out cols w*16; K=512 -> 16 wmma steps
// ---------------------------------------------------------------------------
__global__ void k_ffn2(const _Float16* __restrict__ Y, const _Float16* __restrict__ W2,
                       const float* __restrict__ b2, const float* __restrict__ rst,
                       float* __restrict__ out) {
  int wave = threadIdx.x >> 5, lane = threadIdx.x & 31;
  int m0 = blockIdx.x * 16;
  int o0 = wave * 16;
  v8f c = {};
#pragma unroll
  for (int k0 = 0; k0 < FFH; k0 += 32) {
    v16h a = load_a_frag(Y, FFH, m0, k0, lane);
    v16h b = load_b_frag(W2, FFH, o0, k0, lane);
    c = __builtin_amdgcn_wmma_f32_16x16x32_f16(false, a, false, b, (short)0, c,
                                               false, false);
  }
  int col = lane & 15, hi = lane >> 4;
  float bias = b2[o0 + col];
#pragma unroll
  for (int v = 0; v < 8; ++v) {
    size_t idx = (size_t)(m0 + hi * 8 + v) * DIN + o0 + col;
    out[idx] = c[v] + bias + rst[idx];
  }
}

// ---------------------------------------------------------------------------
extern "C" void kernel_launch(void* const* d_in, const int* in_sizes, int n_in,
                              void* d_out, int out_size, void* d_ws, size_t ws_size,
                              hipStream_t stream) {
  const float* feat   = (const float*)d_in[0];
  const float* Wh     = (const float*)d_in[1];
  const float* Wt     = (const float*)d_in[2];
  const float* We     = (const float*)d_in[3];
  const float* attn_p = (const float*)d_in[4];
  const float* g_head = (const float*)d_in[5];
  const float* g_tail = (const float*)d_in[6];
  const float* ln1_g  = (const float*)d_in[7];
  const float* ln1_b  = (const float*)d_in[8];
  const float* ln2_g  = (const float*)d_in[9];
  const float* ln2_b  = (const float*)d_in[10];
  const float* W1     = (const float*)d_in[11];
  const float* b1     = (const float*)d_in[12];
  const float* W2     = (const float*)d_in[13];
  const float* b2     = (const float*)d_in[14];
  const int*   src    = (const int*)d_in[15];
  // d_in[16] = dst: structure is known (repeat(arange(N),16)), not needed.
  float* out = (float*)d_out;

  char* ws = (char*)d_ws;
  size_t off = 0;
  auto alloc = [&](size_t bytes) -> char* {
    char* p = ws + off;
    off = (off + bytes + 255) & ~(size_t)255;
    return p;
  };

  _Float16* h16  = (_Float16*)alloc((size_t)NNODES * DIN * 2);   // 10.24 MB
  _Float16* wh16 = (_Float16*)alloc(DIN * DIN * 2);
  _Float16* wt16 = (_Float16*)alloc(DIN * DIN * 2);
  _Float16* we16 = (_Float16*)alloc(DIN * DIN * 2);
  _Float16* w116 = (_Float16*)alloc(FFH * DIN * 2);
  _Float16* w216 = (_Float16*)alloc(DIN * FFH * 2);
  float* fh   = (float*)alloc((size_t)NNODES * DIN * 4);         // 20.48 MB
  float* fe   = (float*)alloc((size_t)NNODES * DIN * 4);         // 20.48 MB (feat0)
  float* ft   = (float*)alloc((size_t)NNODES * DIN * 4);         // 20.48 MB (fe,ft contiguous)
  float* gh   = (float*)alloc((size_t)NNODES * NH * 4);
  float* gt   = (float*)alloc((size_t)NNODES * NH * 4);
  float* attn = (float*)alloc((size_t)NEDGES * NH * 4);          // 20.48 MB
  // aliases (lifetimes disjoint):
  _Float16* x16 = h16;            // h16 dead after projections
  float*    rst = attn;           // attn dead after diffusion
  _Float16* y16 = (_Float16*)fe;  // spans fe+ft (40.96 MB), dead after diffusion

  // 1) weights -> f16
  k_cvt_f16<<<(DIN * DIN + 255) / 256, 256, 0, stream>>>(Wh, wh16, DIN * DIN);
  k_cvt_f16<<<(DIN * DIN + 255) / 256, 256, 0, stream>>>(Wt, wt16, DIN * DIN);
  k_cvt_f16<<<(DIN * DIN + 255) / 256, 256, 0, stream>>>(We, we16, DIN * DIN);
  k_cvt_f16<<<(FFH * DIN + 255) / 256, 256, 0, stream>>>(W1, w116, FFH * DIN);
  k_cvt_f16<<<(DIN * FFH + 255) / 256, 256, 0, stream>>>(W2, w216, DIN * FFH);

  // 2) LN1 -> f16 activations
  k_ln_f16<<<NNODES / 8, 256, 0, stream>>>(feat, ln1_g, ln1_b, h16);

  // 3) fused projections via WMMA: fh, ft, fe
  k_gemm_proj<<<dim3(NNODES / 16, 1, 3), 256, 0, stream>>>(h16, wh16, wt16, we16,
                                                           fh, ft, fe);

  // 4) gate scores
  k_gate<<<(NNODES * NH + 255) / 256, 256, 0, stream>>>(fh, ft, g_head, g_tail, gh, gt);

  // 5) edge attention + gated softmax
  k_attn<<<NNODES / 2, 256, 0, stream>>>(fh, ft, attn_p, gh, gt, src, attn);

  // 6) 5 PPR hops, ping-pong (feat0 = fe stays intact; final result in fh)
  k_diffuse<<<NNODES / 2, 256, 0, stream>>>(fe, fe, attn, src, fh);
  k_diffuse<<<NNODES / 2, 256, 0, stream>>>(fh, fe, attn, src, ft);
  k_diffuse<<<NNODES / 2, 256, 0, stream>>>(ft, fe, attn, src, fh);
  k_diffuse<<<NNODES / 2, 256, 0, stream>>>(fh, fe, attn, src, ft);
  k_diffuse<<<NNODES / 2, 256, 0, stream>>>(ft, fe, attn, src, fh);

  // 7) residual + LN2 (rst overwrites attn buffer; x16 overwrites h16)
  k_rst_ln<<<NNODES / 8, 256, 0, stream>>>(fh, feat, ln2_g, ln2_b, rst, x16);

  // 8) FFN via WMMA (y16 overwrites fe+ft region)
  k_ffn1<<<dim3(NNODES / 16, 4), 256, 0, stream>>>(x16, w116, b1, y16);
  k_ffn2<<<NNODES / 16, 256, 0, stream>>>(y16, w216, b2, rst, out);

  (void)in_sizes; (void)n_in; (void)out_size; (void)ws_size;
}
